// MultiHeadAttention_34445637714311
// MI455X (gfx1250) — matrix operand order repaired and checked by reading, not compiled, not run on hardware
//
#include <hip/hip_runtime.h>

// ---------------------------------------------------------------------------
// Types for CDNA5 WMMA (wave32)
// ---------------------------------------------------------------------------
typedef __attribute__((ext_vector_type(16))) __bf16 v16bf;
typedef __attribute__((ext_vector_type(8)))  __bf16 bf16x8;
typedef __attribute__((ext_vector_type(4)))  __bf16 bf16x4;
typedef __attribute__((ext_vector_type(8)))  float  v8f;

union FragU { v16bf v; bf16x8 h[2]; };

static __device__ __forceinline__ __bf16 f2bf(float f) {
  unsigned u = __float_as_uint(f);
  u += 0x7FFFu + ((u >> 16) & 1u);            // round-to-nearest-even
  unsigned short hs = (unsigned short)(u >> 16);
  __bf16 r;
  __builtin_memcpy(&r, &hs, sizeof(r));
  return r;
}

// ---------------------------------------------------------------------------
// Problem constants
// ---------------------------------------------------------------------------
#define BATCH   2
#define SEQ     2048
#define DMODEL  1024
#define HEADS   16
#define DKH     64
#define NEGC    (-1000000000.0f)

// ---------------------------------------------------------------------------
// fp32 -> bf16 conversion
// ---------------------------------------------------------------------------
__global__ __launch_bounds__(256) void cvt_f32_bf16(const float* __restrict__ in,
                                                    __bf16* __restrict__ out, int n) {
  for (int i = blockIdx.x * 256 + threadIdx.x; i < n; i += gridDim.x * 256)
    out[i] = f2bf(in[i]);
}

// ---------------------------------------------------------------------------
// Generic WMMA GEMM:  C[m][n] = sum_k A[m][k] * Brow[n][k]   (i.e. A @ B^T)
//   A : bf16 (or f32 converted while staging) row-major, leading dim lda
//   B : bf16 rows-along-K ("B^T" storage), leading dim ldb
// Epilogues (EPI):
//   0 : Q/K projection  -> bf16, head layout (B,H,S,Dk), +bias
//   1 : V projection    -> bf16, head-transposed (B,H,Dk,S), +bias
//   2 : attention scores-> f32  = mask ? acc*scale : NEG, into p_attn region
//   3 : PV context      -> bf16 (B,S,D) layout
//   4 : final out proj  -> f32 + bias, row-major
// Tiling: 256 thr = 8 waves (4x2). BK=32 per stage, v_wmma_f32_16x16x32_bf16.
// ---------------------------------------------------------------------------
template <int EPI, bool AF32, int BM, int BN>
__global__ __launch_bounds__(256) void wmma_gemm(
    const void* __restrict__ Ap, int lda,
    const __bf16* __restrict__ Bp, int ldb,
    const float* __restrict__ bias,
    void* __restrict__ Out,
    const int* __restrict__ maskp, float scale,
    int M, int N, int K,
    long long strideA, long long strideB) {
  constexpr int BK  = 32;
  constexpr int PAD = 8;
  constexpr int LDK = BK + PAD;          // 40 bf16 per LDS row (80B, 16B-aligned rows)
  constexpr int MI  = BM / (4 * 16);     // m-frags per wave (4 waves in M)
  constexpr int NI  = BN / (2 * 16);     // n-frags per wave (2 waves in N)

  __shared__ __bf16 sA[BM * LDK];
  __shared__ __bf16 sB[BN * LDK];

  const int tid  = threadIdx.x;
  const int lane = tid & 31;
  const int lrow = lane & 15;
  const int half = lane >> 4;
  const int wave = tid >> 5;
  const int wm   = wave >> 1;            // 0..3
  const int wn   = wave & 1;             // 0..1

  const int bh    = blockIdx.z;
  const int mBase = blockIdx.y * BM;
  const int nBase = blockIdx.x * BN;

  const float*  Af = AF32 ? ((const float*)Ap)  + (size_t)bh * strideA : nullptr;
  const __bf16* Ab = AF32 ? nullptr : ((const __bf16*)Ap) + (size_t)bh * strideA;
  const __bf16* Bb = Bp + (size_t)bh * strideB;

  const v8f vzero = {0.f, 0.f, 0.f, 0.f, 0.f, 0.f, 0.f, 0.f};
  v8f acc[MI][NI];
#pragma unroll
  for (int mi = 0; mi < MI; ++mi)
#pragma unroll
    for (int ni = 0; ni < NI; ++ni) acc[mi][ni] = vzero;

  for (int k0 = 0; k0 < K; k0 += BK) {
    // ---- stage A tile (BM x 32) into LDS ----
    if (AF32) {
#pragma unroll
      for (int p = 0; p < BM / 32; ++p) {
        const int r = (tid >> 3) + p * 32;
        const int c = (tid & 7) * 4;
        const float4 f = *(const float4*)(Af + (size_t)(mBase + r) * lda + (k0 + c));
        bf16x4 t;
        t[0] = f2bf(f.x); t[1] = f2bf(f.y); t[2] = f2bf(f.z); t[3] = f2bf(f.w);
        *(bf16x4*)&sA[r * LDK + c] = t;
      }
    } else {
#pragma unroll
      for (int p = 0; p < BM / 64; ++p) {
        const int r = (tid >> 2) + p * 64;
        const int c = (tid & 3) * 8;
        *(bf16x8*)&sA[r * LDK + c] =
            *(const bf16x8*)(Ab + (size_t)(mBase + r) * lda + (k0 + c));
      }
    }
    // ---- stage B tile (BN x 32) into LDS ----
#pragma unroll
    for (int p = 0; p < BN / 64; ++p) {
      const int r = (tid >> 2) + p * 64;
      const int c = (tid & 3) * 8;
      *(bf16x8*)&sB[r * LDK + c] =
          *(const bf16x8*)(Bb + (size_t)(nBase + r) * ldb + (k0 + c));
    }
    __syncthreads();

    // ---- load fragments (ISA 7.12.2 layouts) ----
    FragU fa[MI], fb[NI];
#pragma unroll
    for (int mi = 0; mi < MI; ++mi) {
      const int r = (wm * MI + mi) * 16 + lrow;
      fa[mi].h[0] = *(const bf16x8*)&sA[r * LDK + half * 8];        // K = half*8 .. +7
      fa[mi].h[1] = *(const bf16x8*)&sA[r * LDK + 16 + half * 8];   // K = 16+half*8 ..
    }
#pragma unroll
    for (int ni = 0; ni < NI; ++ni) {
      const int r = (wn * NI + ni) * 16 + lrow;
      fb[ni].h[0] = *(const bf16x8*)&sB[r * LDK + half * 8];        // K = half*8 .. +7
      fb[ni].h[1] = *(const bf16x8*)&sB[r * LDK + 16 + half * 8];   // K = 16+half*8 ..
    }
    // ---- WMMA ----
#pragma unroll
    for (int mi = 0; mi < MI; ++mi)
#pragma unroll
      for (int ni = 0; ni < NI; ++ni)
        acc[mi][ni] = __builtin_amdgcn_wmma_f32_16x16x32_bf16(
            false, fa[mi].v, false, fb[ni].v, (short)0, acc[mi][ni], false, false);
    __syncthreads();
  }

  // ---- epilogue: D layout lane L -> N=L%16 ; VGPR r -> M=r+8*(L/16) ----
  float* Outf = (float*)Out;
  __bf16* Outb = (__bf16*)Out;
  const int mrow0 = mBase + (wm * MI) * 16;
  const int ncol0 = nBase + (wn * NI) * 16;
#pragma unroll
  for (int mi = 0; mi < MI; ++mi) {
#pragma unroll
    for (int ni = 0; ni < NI; ++ni) {
#pragma unroll
      for (int r = 0; r < 8; ++r) {
        const int m = mrow0 + mi * 16 + half * 8 + r;
        const int n = ncol0 + ni * 16 + lrow;
        float v = acc[mi][ni][r];
        if (EPI == 0) {            // Q/K proj: (B,H,S,Dk) bf16
          const int b = m >> 11, s = m & 2047, h = n >> 6, dk = n & 63;
          Outb[((size_t)(b * HEADS + h) * SEQ + s) * DKH + dk] = f2bf(v + bias[n]);
        } else if (EPI == 1) {     // V proj: (B,H,Dk,S) bf16
          const int b = m >> 11, s = m & 2047, h = n >> 6, dk = n & 63;
          Outb[((size_t)(b * HEADS + h) * DKH + dk) * SEQ + s] = f2bf(v + bias[n]);
        } else if (EPI == 2) {     // scores: mask + scale, f32
          const int b = bh >> 4;
          const int mk = maskp[((size_t)b * SEQ + m) * SEQ + n];
          Outf[((size_t)bh * SEQ + m) * SEQ + n] = (mk == 0) ? NEGC : v * scale;
        } else if (EPI == 3) {     // context: (B,S,D) bf16
          const int b = bh >> 4, h = bh & 15;
          Outb[((size_t)(b * SEQ + m)) * DMODEL + h * DKH + n] = f2bf(v);
        } else {                   // final: f32 + bias
          Outf[(size_t)m * N + n] = v + bias[n];
        }
      }
    }
  }
}

// ---------------------------------------------------------------------------
// Row softmax over 2048 columns, in place (f32). One block per row.
// ---------------------------------------------------------------------------
__global__ __launch_bounds__(256) void softmax_rows(float* __restrict__ p) {
  __shared__ float red[256];
  float* rp = p + (size_t)blockIdx.x * SEQ;
  const int tid = threadIdx.x;

  float m = -3.4e38f;
  for (int j = tid; j < SEQ; j += 256) m = fmaxf(m, rp[j]);
  red[tid] = m; __syncthreads();
  for (int s = 128; s > 0; s >>= 1) {
    if (tid < s) red[tid] = fmaxf(red[tid], red[tid + s]);
    __syncthreads();
  }
  const float mx = red[0];
  __syncthreads();

  float sum = 0.f;
  for (int j = tid; j < SEQ; j += 256) {
    const float e = __expf(rp[j] - mx);
    rp[j] = e;
    sum += e;
  }
  red[tid] = sum; __syncthreads();
  for (int s = 128; s > 0; s >>= 1) {
    if (tid < s) red[tid] += red[tid + s];
    __syncthreads();
  }
  const float inv = 1.0f / red[0];
  for (int j = tid; j < SEQ; j += 256) rp[j] *= inv;
}

// ---------------------------------------------------------------------------
// Launch
// ---------------------------------------------------------------------------
extern "C" void kernel_launch(void* const* d_in, const int* in_sizes, int n_in,
                              void* d_out, int out_size, void* d_ws, size_t ws_size,
                              hipStream_t stream) {
  const float* q    = (const float*)d_in[0];
  const float* k    = (const float*)d_in[1];
  const float* v    = (const float*)d_in[2];
  const int*   mask = (const int*)  d_in[3];
  const float* W_q  = (const float*)d_in[4];
  const float* b_q  = (const float*)d_in[5];
  const float* W_k  = (const float*)d_in[6];
  const float* b_k  = (const float*)d_in[7];
  const float* W_v  = (const float*)d_in[8];
  const float* b_v  = (const float*)d_in[9];
  const float* W_o  = (const float*)d_in[10];
  const float* b_o  = (const float*)d_in[11];

  float* out_final = (float*)d_out;                                   // (B,S,D)
  float* p_attn    = out_final + (size_t)BATCH * SEQ * DMODEL;        // (B,H,S,S)

  const size_t NXD = (size_t)BATCH * SEQ * DMODEL;  // 4,194,304
  const size_t NW  = (size_t)DMODEL * DMODEL;       // 1,048,576

  __bf16* ws = (__bf16*)d_ws;
  size_t o = 0;
  __bf16* qb  = ws + o; o += NXD;
  __bf16* kb  = ws + o; o += NXD;
  __bf16* vb  = ws + o; o += NXD;
  __bf16* wqb = ws + o; o += NW;
  __bf16* wkb = ws + o; o += NW;
  __bf16* wvb = ws + o; o += NW;
  __bf16* wob = ws + o; o += NW;
  __bf16* Qh  = ws + o; o += NXD;   // (B,H,S,Dk)
  __bf16* Kh  = ws + o; o += NXD;   // (B,H,S,Dk)
  __bf16* VhT = ws + o; o += NXD;   // (B,H,Dk,S)
  __bf16* ctx = ws + o; o += NXD;   // (B,S,D)

  // 1) convert inputs/weights to bf16
  cvt_f32_bf16<<<4096, 256, 0, stream>>>(q,   qb,  (int)NXD);
  cvt_f32_bf16<<<4096, 256, 0, stream>>>(k,   kb,  (int)NXD);
  cvt_f32_bf16<<<4096, 256, 0, stream>>>(v,   vb,  (int)NXD);
  cvt_f32_bf16<<<1024, 256, 0, stream>>>(W_q, wqb, (int)NW);
  cvt_f32_bf16<<<1024, 256, 0, stream>>>(W_k, wkb, (int)NW);
  cvt_f32_bf16<<<1024, 256, 0, stream>>>(W_v, wvb, (int)NW);
  cvt_f32_bf16<<<1024, 256, 0, stream>>>(W_o, wob, (int)NW);

  // 2) projections: (4096x1024) @ W^T (+bias)
  {
    dim3 grid(DMODEL / 128, (BATCH * SEQ) / 128, 1);
    wmma_gemm<0, false, 128, 128><<<grid, 256, 0, stream>>>(
        qb, DMODEL, wqb, DMODEL, b_q, Qh, nullptr, 1.0f,
        BATCH * SEQ, DMODEL, DMODEL, 0, 0);
    wmma_gemm<0, false, 128, 128><<<grid, 256, 0, stream>>>(
        kb, DMODEL, wkb, DMODEL, b_k, Kh, nullptr, 1.0f,
        BATCH * SEQ, DMODEL, DMODEL, 0, 0);
    wmma_gemm<1, false, 128, 128><<<grid, 256, 0, stream>>>(
        vb, DMODEL, wvb, DMODEL, b_v, VhT, nullptr, 1.0f,
        BATCH * SEQ, DMODEL, DMODEL, 0, 0);
  }

  // 3) scores = Qh @ Kh^T * (1/sqrt(Dk)), masked -> f32 p_attn region
  {
    dim3 grid(SEQ / 128, SEQ / 128, BATCH * HEADS);
    wmma_gemm<2, false, 128, 128><<<grid, 256, 0, stream>>>(
        Qh, DKH, Kh, DKH, nullptr, p_attn, mask, 0.125f,
        SEQ, SEQ, DKH, (long long)SEQ * DKH, (long long)SEQ * DKH);
  }

  // 4) softmax rows (in place on p_attn)
  softmax_rows<<<BATCH * HEADS * SEQ, 256, 0, stream>>>(p_attn);

  // 5) context = p_attn(f32) @ Vh  (B-operand = VhT rows along K)
  {
    dim3 grid(1, SEQ / 128, BATCH * HEADS);
    wmma_gemm<3, true, 128, 64><<<grid, 256, 0, stream>>>(
        p_attn, SEQ, VhT, SEQ, nullptr, ctx, nullptr, 1.0f,
        SEQ, DKH, SEQ, (long long)SEQ * SEQ, (long long)DKH * SEQ);
  }

  // 6) final = ctx @ W_o^T + b_o -> f32 out
  {
    dim3 grid(DMODEL / 128, (BATCH * SEQ) / 128, 1);
    wmma_gemm<4, false, 128, 128><<<grid, 256, 0, stream>>>(
        ctx, DMODEL, wob, DMODEL, b_o, out_final, nullptr, 1.0f,
        BATCH * SEQ, DMODEL, DMODEL, 0, 0);
  }
}
